// RoPEMaskedAttentionHead_45243185496367
// MI455X (gfx1250) — matrix operand order repaired and checked by reading, not compiled, not run on hardware
//
#include <hip/hip_runtime.h>

typedef __attribute__((ext_vector_type(16))) _Float16 v16h;
typedef __attribute__((ext_vector_type(8)))  float    v8f;

union AFrag { v16h v; uint4 u[2]; };
union H8 { uint4 u; _Float16 h[8]; };
union H4 { uint2 u; _Float16 h[4]; };

#define D_MODEL 4096
#define D_HEAD  128
#define SEQ     4096
#define BATCH   4

__device__ inline v8f zero8() {
  v8f z = {0.f, 0.f, 0.f, 0.f, 0.f, 0.f, 0.f, 0.f};
  return z;
}

// Async copy 16B global -> LDS (per lane), tracked by ASYNCcnt.
__device__ inline void async_copy_b128(unsigned lds_off, const void* gsrc) {
  asm volatile("global_load_async_to_lds_b128 %0, %1, off"
               :: "v"(lds_off), "v"(gsrc) : "memory");
}
__device__ inline void wait_async0() {
  asm volatile("s_wait_asynccnt 0x0" ::: "memory");
}

// ---------------- kernel 0: fp32 -> fp16 weight convert ----------------
__global__ __launch_bounds__(256) void cvt_f32_f16(const float* __restrict__ src,
                                                   _Float16* __restrict__ dst, int n) {
  int i = (blockIdx.x * 256 + threadIdx.x) * 4;
  if (i + 3 < n) {
    float4 f = *reinterpret_cast<const float4*>(src + i);
    H4 o;
    o.h[0] = (_Float16)f.x; o.h[1] = (_Float16)f.y;
    o.h[2] = (_Float16)f.z; o.h[3] = (_Float16)f.w;
    *reinterpret_cast<uint2*>(dst + i) = o.u;
  }
}

// ---------------- kernel 1: QKV projection + bias + RoPE ----------------
// Block: 32 flat rows of x, 256 threads = 8 waves.
// Wave w: rows (w>>2)*16, col-tiles (w&3)*6 .. +5 of the 24 (q|k|v) 16-wide tiles.
__global__ __launch_bounds__(256) void qkv_rope(
    const float* __restrict__ x,
    const _Float16* __restrict__ wh,   // [3][128][4096] f16, contiguous
    const float* __restrict__ b_q, const float* __restrict__ b_k, const float* __restrict__ b_v,
    const float* __restrict__ sinb, const float* __restrict__ cosb,
    _Float16* __restrict__ q_rot, _Float16* __restrict__ k_rot, _Float16* __restrict__ v_out)
{
  __shared__ _Float16 xs[32 * 40];   // 32 rows x 32 K, stride 40 (pad)

  const int t     = threadIdx.x;
  const int lane  = t & 31;
  const int wave  = t >> 5;
  const int lanen = lane & 15;
  const int hi8   = (lane & 16) >> 1;     // 0 or 8
  const int rb    = blockIdx.x * 32;      // flat row base (b*4096+s)
  const int rowg  = (wave >> 2) * 16;
  const int quad  = wave & 3;

  v8f acc[6];
  #pragma unroll
  for (int j = 0; j < 6; ++j) acc[j] = zero8();

  const int sr = t >> 3;           // staging row 0..31
  const int sc4 = (t & 7) * 4;     // staging col 0,4,...,28
  const float* xstage = x + (size_t)(rb + sr) * D_MODEL + sc4;

  // per-tile weight base pointers (per-lane column)
  const _Float16* wbase[6];
  #pragma unroll
  for (int j = 0; j < 6; ++j) {
    const int ctg  = quad * 6 + j;
    const int mat  = ctg >> 3;
    const int hcol = (ctg & 7) * 16 + lanen;
    wbase[j] = wh + (size_t)mat * ((size_t)D_HEAD * D_MODEL)
                  + (size_t)hcol * D_MODEL + (lane & 16);
  }

  for (int k0 = 0; k0 < D_MODEL; k0 += 32) {
    __syncthreads();
    float4 f0 = *reinterpret_cast<const float4*>(xstage + k0);
    H4 pk;
    pk.h[0] = (_Float16)f0.x; pk.h[1] = (_Float16)f0.y;
    pk.h[2] = (_Float16)f0.z; pk.h[3] = (_Float16)f0.w;
    *reinterpret_cast<uint2*>(&xs[sr * 40 + sc4]) = pk.u;
    if (k0 + 32 < D_MODEL) __builtin_prefetch(xstage + k0 + 32, 0, 1);
    __syncthreads();

    // A fragment (16x32 f16): lane row = lanen, K = 2*hi8 .. 2*hi8+15 contiguous
    AFrag a;
    const int arow = rowg + lanen;
    a.u[0] = *reinterpret_cast<const uint4*>(&xs[arow * 40 + 2 * hi8]);
    a.u[1] = *reinterpret_cast<const uint4*>(&xs[arow * 40 + 2 * hi8 + 8]);

    // pre-load all 6 B fragments, then run the 6 WMMAs back-to-back
    AFrag bf[6];
    #pragma unroll
    for (int j = 0; j < 6; ++j) {
      const _Float16* wp = wbase[j] + k0;
      bf[j].u[0] = *reinterpret_cast<const uint4*>(wp);
      bf[j].u[1] = *reinterpret_cast<const uint4*>(wp + 8);
    }
    #pragma unroll
    for (int j = 0; j < 6; ++j)
      acc[j] = __builtin_amdgcn_wmma_f32_16x16x32_f16(false, a.v, false, bf[j].v,
                                                      (short)0, acc[j], false, false);
  }

  // Epilogue: bias + RoPE (exactly as reference buffers define) + f16 store
  const float* biases[3] = { b_q, b_k, b_v };
  _Float16* dsts[3] = { q_rot, k_rot, v_out };
  for (int j = 0; j < 6; ++j) {
    const int ctg  = quad * 6 + j;
    const int mat  = ctg >> 3;
    const int hcol = (ctg & 7) * 16 + lanen;
    const float bv = biases[mat][hcol];
    _Float16* dst  = dsts[mat];
    #pragma unroll
    for (int r = 0; r < 8; ++r) {
      const int gr = rb + rowg + r + hi8;   // flat row
      float val = acc[j][r] + bv;
      if (mat < 2) {
        const int s = gr & (SEQ - 1);
        const float c  = cosb[s * D_HEAD + hcol];
        const float sn = sinb[s * D_HEAD + hcol];
        const float other = __shfl_xor(val, 1, 32);     // pair partner (hcol^1)
        const float vsw = (hcol & 1) ? val : other;     // value at odd index of pair
        val = val * c + vsw * sn;
      }
      dst[(size_t)gr * D_HEAD + hcol] = (_Float16)val;
    }
  }
}

// ---------------- kernel 2: fused causal attention (flash-style) ----------------
// Block: (query tile of 128, batch). 8 waves x 16 query rows. Key tiles of 64.
#define PK 136   // padded LDS stride for K tile rows (272B -> bank step 4)
__global__ __launch_bounds__(256) void attn(
    const _Float16* __restrict__ q_rot, const _Float16* __restrict__ k_rot,
    const _Float16* __restrict__ v_h, float* __restrict__ out)
{
  __shared__ _Float16 vT[128 * 72];       // [head dim][key] padded stride 72
  __shared__ _Float16 pP[8 * 16 * 72];    // per-wave P tile [16 rows][64 keys]
  __shared__ _Float16 kT[64 * PK];        // K tile, natural [key][head dim], padded

  const int t     = threadIdx.x;
  const int lane  = t & 31;
  const int wave  = t >> 5;
  const int lanen = lane & 15;
  const int hi8   = (lane & 16) >> 1;
  const int b     = blockIdx.y;
  const int qt    = blockIdx.x;
  const int qbase = qt * 128 + wave * 16;
  const float scale = 0.08838834764831845f;  // 1/sqrt(128)

  // Q fragments: 4 K-chunks of 32 over head dim
  AFrag qf[4];
  {
    const _Float16* qp = q_rot + ((size_t)b * SEQ + qbase + lanen) * D_HEAD;
    #pragma unroll
    for (int kc = 0; kc < 4; ++kc) {
      qf[kc].u[0] = *reinterpret_cast<const uint4*>(qp + kc * 32 + 2 * hi8);
      qf[kc].u[1] = *reinterpret_cast<const uint4*>(qp + kc * 32 + 2 * hi8 + 8);
    }
  }

  v8f O[8];
  #pragma unroll
  for (int j = 0; j < 8; ++j) O[j] = zero8();
  float mrow[8], lrow[8];
  #pragma unroll
  for (int r = 0; r < 8; ++r) { mrow[r] = -3.0e38f; lrow[r] = 0.f; }

  _Float16* pw = &pP[wave * 16 * 72];
  const int ntiles = qt * 2 + 2;   // causal: keys up to 64*ntiles cover diagonal

  for (int kt = 0; kt < ntiles; ++kt) {
    const int kb = kt * 64;
    __syncthreads();

    // --- async-stage K tile (natural layout, padded) into LDS ---
    #pragma unroll
    for (int i = 0; i < 4; ++i) {
      const int li  = i * 256 + t;          // 1024 x 16B chunks = 64 rows x 256B
      const int row = li >> 4;
      const int c8  = (li & 15) * 8;        // halves
      const unsigned loff =
          (unsigned)(size_t)&kT[row * PK + c8];
      async_copy_b128(loff, k_rot + ((size_t)b * SEQ + kb + row) * D_HEAD + c8);
    }

    // --- stage V tile transposed (VALU): vT[dim][key] ---
    #pragma unroll
    for (int i = 0; i < 4; ++i) {
      const int li  = i * 256 + t;
      const int key = li >> 4;
      const int d8  = (li & 15) * 8;
      H8 pv;
      pv.u = *reinterpret_cast<const uint4*>(
          v_h + ((size_t)b * SEQ + kb + key) * D_HEAD + d8);
      #pragma unroll
      for (int jj = 0; jj < 8; ++jj) vT[(d8 + jj) * 72 + key] = pv.h[jj];
    }
    wait_async0();
    __syncthreads();

    // --- S = Q K^T : 4 key col-tiles x 4 K-chunks, B frags from LDS ---
    v8f sc[4];
    #pragma unroll
    for (int ct = 0; ct < 4; ++ct) sc[ct] = zero8();
    #pragma unroll
    for (int ct = 0; ct < 4; ++ct) {
      const _Float16* kp = &kT[(ct * 16 + lanen) * PK + (lane & 16)];
      AFrag bf[4];
      #pragma unroll
      for (int kc = 0; kc < 4; ++kc) {
        bf[kc].u[0] = *reinterpret_cast<const uint4*>(kp + kc * 32);
        bf[kc].u[1] = *reinterpret_cast<const uint4*>(kp + kc * 32 + 8);
      }
      #pragma unroll
      for (int kc = 0; kc < 4; ++kc)
        sc[ct] = __builtin_amdgcn_wmma_f32_16x16x32_f16(false, qf[kc].v, false, bf[kc].v,
                                                        (short)0, sc[ct], false, false);
    }

    // --- causal mask + online softmax (fp32) ---
    #pragma unroll
    for (int r = 0; r < 8; ++r) {
      const int qi = qbase + r + hi8;
      float mx = -3.0e38f;
      #pragma unroll
      for (int ct = 0; ct < 4; ++ct) {
        const int kj = kb + ct * 16 + lanen;
        float sv = sc[ct][r] * scale;
        sv = (kj > qi) ? -3.0e38f : sv;
        sc[ct][r] = sv;
        mx = fmaxf(mx, sv);
      }
      for (int off = 1; off < 16; off <<= 1) mx = fmaxf(mx, __shfl_xor(mx, off, 32));
      const float mnew  = fmaxf(mrow[r], mx);
      const float alpha = __expf(mrow[r] - mnew);
      mrow[r] = mnew;
      float rs = 0.f;
      #pragma unroll
      for (int ct = 0; ct < 4; ++ct) {
        const float p = __expf(sc[ct][r] - mnew);
        sc[ct][r] = p;
        rs += p;
      }
      for (int off = 1; off < 16; off <<= 1) rs += __shfl_xor(rs, off, 32);
      lrow[r] = lrow[r] * alpha + rs;
      #pragma unroll
      for (int j = 0; j < 8; ++j) O[j][r] *= alpha;
    }

    // --- P (C-layout fp32) -> per-wave LDS f16 tile [16][64] ---
    #pragma unroll
    for (int ct = 0; ct < 4; ++ct)
      #pragma unroll
      for (int r = 0; r < 8; ++r)
        pw[(r + hi8) * 72 + ct * 16 + lanen] = (_Float16)sc[ct][r];

    // --- O += P * V  (K-dim = 64 keys, two 32-chunks) ---
    #pragma unroll
    for (int kc2 = 0; kc2 < 2; ++kc2) {
      AFrag pf;
      pf.u[0] = *reinterpret_cast<const uint4*>(&pw[lanen * 72 + kc2 * 32 + 2 * hi8]);
      pf.u[1] = *reinterpret_cast<const uint4*>(&pw[lanen * 72 + kc2 * 32 + 2 * hi8 + 8]);
      #pragma unroll
      for (int j = 0; j < 8; ++j) {
        const _Float16* vp = &vT[(j * 16 + lanen) * 72 + kc2 * 32 + (lane & 16)];
        AFrag vf;
        vf.u[0] = *reinterpret_cast<const uint4*>(vp);
        vf.u[1] = *reinterpret_cast<const uint4*>(vp + 8);
        O[j] = __builtin_amdgcn_wmma_f32_16x16x32_f16(false, pf.v, false, vf.v,
                                                      (short)0, O[j], false, false);
      }
    }
  }

  // epilogue: normalize and store fp32
  #pragma unroll
  for (int j = 0; j < 8; ++j) {
    #pragma unroll
    for (int r = 0; r < 8; ++r) {
      const int row = qbase + r + hi8;
      out[((size_t)b * SEQ + row) * D_HEAD + j * 16 + lanen] = O[j][r] / lrow[r];
    }
  }
}

// ---------------- launcher ----------------
extern "C" void kernel_launch(void* const* d_in, const int* in_sizes, int n_in,
                              void* d_out, int out_size, void* d_ws, size_t ws_size,
                              hipStream_t stream) {
  (void)in_sizes; (void)n_in; (void)out_size; (void)ws_size;
  const float* x    = (const float*)d_in[0];
  const float* w_q  = (const float*)d_in[1];
  const float* b_q  = (const float*)d_in[2];
  const float* w_k  = (const float*)d_in[3];
  const float* b_k  = (const float*)d_in[4];
  const float* w_v  = (const float*)d_in[5];
  const float* b_v  = (const float*)d_in[6];
  const float* sinb = (const float*)d_in[7];
  const float* cosb = (const float*)d_in[8];
  float* out = (float*)d_out;

  _Float16* wsh = (_Float16*)d_ws;
  const size_t WN = (size_t)D_HEAD * D_MODEL;       // 524288 per weight matrix
  const size_t QN = (size_t)BATCH * SEQ * D_HEAD;   // 2097152 per activation
  _Float16* wh    = wsh;                 // 3*WN halves
  _Float16* q_rot = wsh + 3 * WN;
  _Float16* k_rot = q_rot + QN;
  _Float16* v_h   = k_rot + QN;

  cvt_f32_f16<<<(int)(WN / 1024), 256, 0, stream>>>(w_q, wh, (int)WN);
  cvt_f32_f16<<<(int)(WN / 1024), 256, 0, stream>>>(w_k, wh + WN, (int)WN);
  cvt_f32_f16<<<(int)(WN / 1024), 256, 0, stream>>>(w_v, wh + 2 * WN, (int)WN);

  qkv_rope<<<(BATCH * SEQ) / 32, 256, 0, stream>>>(x, wh, b_q, b_k, b_v,
                                                   sinb, cosb, q_rot, k_rot, v_h);

  attn<<<dim3(SEQ / 128, BATCH), 256, 0, stream>>>(q_rot, k_rot, v_h, out);
}
